// PhysicsGuidedGCN_52304111730902
// MI455X (gfx1250) — matrix operand order repaired and checked by reading, not compiled, not run on hardware
//
#include <hip/hip_runtime.h>
#include <hip/hip_bf16.h>
#include <math.h>

typedef __attribute__((ext_vector_type(16))) __bf16 v16bf;
typedef __attribute__((ext_vector_type(8)))  float  v8f;

#define B_  4
#define N_  256
#define T_  128
#define F_  64
#define K_  8
#define H_  256
#define E_  8

// ---------------- workspace layout (bytes) ----------------
static constexpr size_t WS_CTX   = 0;                                   // B*F f32      (1 KB)
static constexpr size_t WS_ALPHA = 1024;                                // B*K f32      (128 B)
static constexpr size_t WS_INWT  = 1536;                                // 64*256 bf16  (32 KB)
static constexpr size_t WS_OUTWT = WS_INWT + (size_t)F_*H_*2;           // 256*256 bf16 (128 KB)
static constexpr size_t WS_AS    = WS_OUTWT + (size_t)H_*H_*2;          // B*K*N*N bf16 (4 MB)
static constexpr size_t WS_HT    = WS_AS + (size_t)B_*K_*N_*N_*2;       // B*T*H*N bf16 (64 MB)

union OpA { v16bf v; uint4 q[2]; };

__device__ __forceinline__ unsigned short f2bf(float f) {
  __bf16 h = (__bf16)f;
  return __builtin_bit_cast(unsigned short, h);
}
__device__ __forceinline__ float bf2f(unsigned short u) {
  __bf16 h = __builtin_bit_cast(__bf16, u);
  return (float)h;
}
__device__ __forceinline__ float gelu_f(float x) {
  return 0.5f * x * (1.0f + erff(x * 0.70710678118654752f));
}
__device__ __forceinline__ v8f wmma_bf16(v16bf a, v16bf b, v8f c) {
  return __builtin_amdgcn_wmma_f32_16x16x32_bf16(false, a, false, b, (short)0, c, false, false);
}

// ---------------- K1a: ctx[b,f] = mean over (n,t) of x ----------------
__global__ void k_ctx(const float* __restrict__ x, float* __restrict__ ctx) {
  int b = blockIdx.x >> 6;          // 64 feature-blocks per batch
  int f = blockIdx.x & 63;
  float s = 0.f;
  for (int idx = threadIdx.x; idx < N_ * T_; idx += 256) {
    int n = idx >> 7, t = idx & 127;
    s += x[((size_t)(b * N_ + n) * T_ + t) * F_ + f];
  }
  __shared__ float red[256];
  red[threadIdx.x] = s; __syncthreads();
  for (int o = 128; o > 0; o >>= 1) {
    if (threadIdx.x < (unsigned)o) red[threadIdx.x] += red[threadIdx.x + o];
    __syncthreads();
  }
  if (threadIdx.x == 0) ctx[b * F_ + f] = red[0] * (1.0f / (N_ * T_));
}

// ---------------- K1b: tiny gating MLPs -> alpha[b,k] ----------------
__global__ void k_alpha(const float* __restrict__ ctx,
                        const float* __restrict__ cw1, const float* __restrict__ cb1,
                        const float* __restrict__ cw2, const float* __restrict__ cb2,
                        const float* __restrict__ lag,
                        const float* __restrict__ gw1, const float* __restrict__ gb1,
                        const float* __restrict__ gw2, const float* __restrict__ gb2,
                        float* __restrict__ alpha) {
  int tid = threadIdx.x;
  if (tid >= B_ * K_) return;
  int b = tid / K_, k = tid % K_;
  float c1[E_], cf[E_], g1[E_], gi[2 * E_];
  for (int e = 0; e < E_; e++) {
    float s = cb1[e];
    for (int f = 0; f < F_; f++) s += ctx[b * F_ + f] * cw1[f * E_ + e];
    c1[e] = gelu_f(s);
  }
  for (int e = 0; e < E_; e++) {
    float s = cb2[e];
    for (int i = 0; i < E_; i++) s += c1[i] * cw2[i * E_ + e];
    cf[e] = s;
  }
  for (int e = 0; e < E_; e++) { gi[e] = lag[k * E_ + e]; gi[E_ + e] = cf[e]; }
  for (int e = 0; e < E_; e++) {
    float s = gb1[e];
    for (int i = 0; i < 2 * E_; i++) s += gi[i] * gw1[i * E_ + e];
    g1[e] = gelu_f(s);
  }
  float v = gb2[0];
  for (int e = 0; e < E_; e++) v += g1[e] * gw2[e];
  alpha[tid] = 1.0f / (1.0f + expf(-v));
}

// ---------------- K1c: transpose weights -> bf16 (B-operand layout) ----------------
__global__ void k_pack(const float* __restrict__ in_w, const float* __restrict__ out_w,
                       unsigned short* __restrict__ inwT, unsigned short* __restrict__ outwT) {
  int i = blockIdx.x * 256 + threadIdx.x;
  if (i < F_ * H_) { int c = i / F_, k = i % F_; inwT[i]  = f2bf(in_w[k * H_ + c]); }
  if (i < H_ * H_) { int c = i / H_, k = i % H_; outwT[i] = f2bf(out_w[k * H_ + c]); }
}

// ---------------- K1d: A_s[b,k,i,j] = alpha[b,k] * A[k,i,j]/rowsum (bf16) ----------------
__global__ void k_As(const float* __restrict__ A, const float* __restrict__ alpha,
                     unsigned short* __restrict__ As) {
  int k = blockIdx.x >> 8;
  int i = blockIdx.x & 255;
  int j = threadIdx.x;
  float a = A[((size_t)k * N_ + i) * N_ + j];
  __shared__ float red[256];
  red[j] = a; __syncthreads();
  for (int o = 128; o > 0; o >>= 1) {
    if (j < o) red[j] += red[j + o];
    __syncthreads();
  }
  float an = a / fmaxf(red[0], 1e-8f);
  for (int b = 0; b < B_; b++)
    As[(((size_t)b * K_ + k) * N_ + i) * N_ + j] = f2bf(alpha[b * K_ + k] * an);
}

// ---------------- K2: h = x @ in_w + in_b, stored transposed hT[b][t][hh][n] bf16 ----------------
__global__ void __launch_bounds__(512)
k_h(const float* __restrict__ x, const unsigned short* __restrict__ inwT,
    const float* __restrict__ in_b, unsigned short* __restrict__ hT) {
  int qi = blockIdx.x & 3;
  int bt = blockIdx.x >> 2;
  int t = bt % T_, b = bt / T_;
  int w = threadIdx.x >> 5, l = threadIdx.x & 31;
  int ti = w >> 2, tcq = w & 3;
  int lrow = l & 15, hi = l >> 4;
  int n0 = qi * 64 + ti * 16;
  const float* xrow = x + ((size_t)(b * N_ + n0 + lrow) * T_ + t) * F_;

  v8f zero = {0.f, 0.f, 0.f, 0.f, 0.f, 0.f, 0.f, 0.f};
  v8f acc[4] = {zero, zero, zero, zero};

  for (int kc = 0; kc < F_; kc += 32) {
    int ka = kc + hi * 8;
    float4 f0 = *(const float4*)(xrow + ka);
    float4 f1 = *(const float4*)(xrow + ka + 4);
    float4 f2 = *(const float4*)(xrow + ka + 16);
    float4 f3 = *(const float4*)(xrow + ka + 20);
    v16bf av;
    av[0]=(__bf16)f0.x; av[1]=(__bf16)f0.y; av[2]=(__bf16)f0.z; av[3]=(__bf16)f0.w;
    av[4]=(__bf16)f1.x; av[5]=(__bf16)f1.y; av[6]=(__bf16)f1.z; av[7]=(__bf16)f1.w;
    av[8]=(__bf16)f2.x; av[9]=(__bf16)f2.y; av[10]=(__bf16)f2.z; av[11]=(__bf16)f2.w;
    av[12]=(__bf16)f3.x; av[13]=(__bf16)f3.y; av[14]=(__bf16)f3.z; av[15]=(__bf16)f3.w;
#pragma unroll
    for (int ct = 0; ct < 4; ct++) {
      int c = tcq * 64 + ct * 16 + lrow;
      OpA bop;
      const uint4* p = (const uint4*)(inwT + (size_t)c * F_ + kc + hi * 8);
      bop.q[0] = p[0]; bop.q[1] = p[2];
      acc[ct] = wmma_bf16(av, bop.v, acc[ct]);
    }
  }
#pragma unroll
  for (int ct = 0; ct < 4; ct++) {
    int c = tcq * 64 + ct * 16 + lrow;
    float bias = in_b[c];
    unsigned u0 = (unsigned)f2bf(acc[ct][0] + bias) | ((unsigned)f2bf(acc[ct][1] + bias) << 16);
    unsigned u1 = (unsigned)f2bf(acc[ct][2] + bias) | ((unsigned)f2bf(acc[ct][3] + bias) << 16);
    unsigned u2 = (unsigned)f2bf(acc[ct][4] + bias) | ((unsigned)f2bf(acc[ct][5] + bias) << 16);
    unsigned u3 = (unsigned)f2bf(acc[ct][6] + bias) | ((unsigned)f2bf(acc[ct][7] + bias) << 16);
    uint4 pk; pk.x = u0; pk.y = u1; pk.z = u2; pk.w = u3;
    *(uint4*)(hT + ((size_t)(b * T_ + t) * H_ + c) * N_ + n0 + hi * 8) = pk;
  }
}

// ---------------- K3: lag aggregation + H x H projection + residual + LN + GELU ----------------
__global__ void __launch_bounds__(512)
k_main(const unsigned short* __restrict__ As, const unsigned short* __restrict__ hT,
       const unsigned short* __restrict__ outwT, const float* __restrict__ out_b,
       const float* __restrict__ ln_g, const float* __restrict__ ln_b,
       float* __restrict__ out) {
  __shared__ unsigned short agg[64 * 264];   // 64 rows x 256 cols bf16, padded stride
  __shared__ float lnp_s[64][4];
  __shared__ float lnp_q[64][4];

  int qi = blockIdx.x & 3;
  int bt = blockIdx.x >> 2;
  int t = bt % T_, b = bt / T_;
  int w = threadIdx.x >> 5, l = threadIdx.x & 31;
  int ti = w >> 2, tcq = w & 3;
  int lrow = l & 15, hi = l >> 4;
  int i0l = ti * 16;
  int i0 = qi * 64 + i0l;

  v8f zero = {0.f, 0.f, 0.f, 0.f, 0.f, 0.f, 0.f, 0.f};
  v8f acc[4] = {zero, zero, zero, zero};

  // ---- lag-graph aggregation: acc = sum_k (alpha*A_norm)[b,k] @ hT[b,t-k] ----
  const unsigned short* Asb = As + (size_t)b * K_ * N_ * N_;
  for (int k = 0; k < K_; k++) {
    int ts = t - k;
    if (ts < 0) break;                       // zero padding (uniform across block)
    const unsigned short* arow  = Asb + ((size_t)k * N_ + i0 + lrow) * N_;
    const unsigned short* hbase = hT + (size_t)(b * T_ + ts) * H_ * N_;
#pragma unroll
    for (int jc = 0; jc < 8; jc++) {
      int j0 = jc * 32;
      OpA a;
      a.q[0] = *(const uint4*)(arow + j0 + hi * 8);
      a.q[1] = *(const uint4*)(arow + j0 + 16 + hi * 8);
#pragma unroll
      for (int ct = 0; ct < 4; ct++) {
        int c = tcq * 64 + ct * 16 + lrow;
        OpA bb;
        const uint4* p = (const uint4*)(hbase + (size_t)c * N_ + j0 + hi * 8);
        bb.q[0] = p[0]; bb.q[1] = p[2];
        acc[ct] = wmma_bf16(a.v, bb.v, acc[ct]);
      }
    }
  }

  // ---- stage agg to LDS (bf16, A-operand layout: row-major over hh) ----
#pragma unroll
  for (int ct = 0; ct < 4; ct++) {
    int c = tcq * 64 + ct * 16 + lrow;
#pragma unroll
    for (int r = 0; r < 8; r++)
      agg[(i0l + hi * 8 + r) * 264 + c] = f2bf(acc[ct][r]);
  }
  __syncthreads();

  // ---- projection: accp = agg @ out_w ----
  v8f accp[4] = {zero, zero, zero, zero};
  const unsigned short* aggrow = agg + (size_t)(i0l + lrow) * 264;
#pragma unroll
  for (int kc = 0; kc < 8; kc++) {
    int k0 = kc * 32;
    OpA a;
    a.q[0] = *(const uint4*)(aggrow + k0 + hi * 8);
    a.q[1] = *(const uint4*)(aggrow + k0 + 16 + hi * 8);
#pragma unroll
    for (int ct = 0; ct < 4; ct++) {
      int c = tcq * 64 + ct * 16 + lrow;
      OpA bb;
      const uint4* p = (const uint4*)(outwT + (size_t)c * H_ + k0 + hi * 8);
      bb.q[0] = p[0]; bb.q[1] = p[2];
      accp[ct] = wmma_bf16(a.v, bb.v, accp[ct]);
    }
  }

  // ---- bias + residual + per-row partial sums ----
  float ps[8] = {0,0,0,0,0,0,0,0};
  float pq[8] = {0,0,0,0,0,0,0,0};
#pragma unroll
  for (int ct = 0; ct < 4; ct++) {
    int c = tcq * 64 + ct * 16 + lrow;
    float ob = out_b[c];
    uint4 rq = *(const uint4*)(hT + ((size_t)(b * T_ + t) * H_ + c) * N_ + i0 + hi * 8);
    const unsigned short* ru = (const unsigned short*)&rq;
#pragma unroll
    for (int r = 0; r < 8; r++) {
      float v = accp[ct][r] + ob + bf2f(ru[r]);
      accp[ct][r] = v;
      ps[r] += v; pq[r] += v * v;
    }
  }
  // reduce over the 16 columns held by each lane-half
#pragma unroll
  for (int m = 1; m < 16; m <<= 1) {
#pragma unroll
    for (int r = 0; r < 8; r++) {
      ps[r] += __shfl_xor(ps[r], m, 32);
      pq[r] += __shfl_xor(pq[r], m, 32);
    }
  }
  if (lrow == 0) {
#pragma unroll
    for (int r = 0; r < 8; r++) {
      lnp_s[i0l + hi * 8 + r][tcq] = ps[r];
      lnp_q[i0l + hi * 8 + r][tcq] = pq[r];
    }
  }
  __syncthreads();

  // ---- LayerNorm + GELU + transposed store out[b][n][t][hh] ----
  float mu[8], rs[8];
#pragma unroll
  for (int r = 0; r < 8; r++) {
    int row = i0l + hi * 8 + r;
    float s1 = lnp_s[row][0] + lnp_s[row][1] + lnp_s[row][2] + lnp_s[row][3];
    float s2 = lnp_q[row][0] + lnp_q[row][1] + lnp_q[row][2] + lnp_q[row][3];
    float m_ = s1 * (1.0f / H_);
    float v_ = s2 * (1.0f / H_) - m_ * m_;
    mu[r] = m_;
    rs[r] = rsqrtf(v_ + 1e-5f);
  }
#pragma unroll
  for (int ct = 0; ct < 4; ct++) {
    int c = tcq * 64 + ct * 16 + lrow;
    float g = ln_g[c], be = ln_b[c];
#pragma unroll
    for (int r = 0; r < 8; r++) {
      float v = (accp[ct][r] - mu[r]) * rs[r] * g + be;
      v = gelu_f(v);
      int i = i0 + hi * 8 + r;
      out[((size_t)(b * N_ + i) * T_ + t) * H_ + c] = v;
    }
  }
}

extern "C" void kernel_launch(void* const* d_in, const int* in_sizes, int n_in,
                              void* d_out, int out_size, void* d_ws, size_t ws_size,
                              hipStream_t stream) {
  (void)in_sizes; (void)n_in; (void)out_size; (void)ws_size;
  const float* x     = (const float*)d_in[0];
  const float* A     = (const float*)d_in[1];
  const float* in_w  = (const float*)d_in[2];
  const float* in_b  = (const float*)d_in[3];
  const float* out_w = (const float*)d_in[4];
  const float* out_b = (const float*)d_in[5];
  const float* lag   = (const float*)d_in[6];
  const float* cw1   = (const float*)d_in[7];
  const float* cb1   = (const float*)d_in[8];
  const float* cw2   = (const float*)d_in[9];
  const float* cb2   = (const float*)d_in[10];
  const float* gw1   = (const float*)d_in[11];
  const float* gb1   = (const float*)d_in[12];
  const float* gw2   = (const float*)d_in[13];
  const float* gb2   = (const float*)d_in[14];
  const float* ln_g  = (const float*)d_in[15];
  const float* ln_b  = (const float*)d_in[16];

  char* ws = (char*)d_ws;
  float*          ctx   = (float*)(ws + WS_CTX);
  float*          alpha = (float*)(ws + WS_ALPHA);
  unsigned short* inwT  = (unsigned short*)(ws + WS_INWT);
  unsigned short* outwT = (unsigned short*)(ws + WS_OUTWT);
  unsigned short* As    = (unsigned short*)(ws + WS_AS);
  unsigned short* hT    = (unsigned short*)(ws + WS_HT);
  float* out = (float*)d_out;

  k_ctx  <<<B_ * F_,           256, 0, stream>>>(x, ctx);
  k_alpha<<<1,                 32,  0, stream>>>(ctx, cw1, cb1, cw2, cb2, lag, gw1, gb1, gw2, gb2, alpha);
  k_pack <<<(H_ * H_) / 256,   256, 0, stream>>>(in_w, out_w, inwT, outwT);
  k_As   <<<K_ * N_,           256, 0, stream>>>(A, alpha, As);
  k_h    <<<B_ * T_ * 4,       512, 0, stream>>>(x, inwT, in_b, hT);
  k_main <<<B_ * T_ * 4,       512, 0, stream>>>(As, hT, outwT, out_b, ln_g, ln_b, out);
}
